// Block_SA_29463475650615
// MI455X (gfx1250) — matrix operand order repaired and checked by reading, not compiled, not run on hardware
//
#include <hip/hip_runtime.h>
#include <stdint.h>

typedef _Float16 f16;
typedef __attribute__((ext_vector_type(16))) _Float16 v16h;
typedef __attribute__((ext_vector_type(8)))  float    v8f;

#define DIM   320
#define HEADS 32
#define DH    10
#define MTOK  131072   // 8 * 128 * 128 tokens

union Frag { v16h h; uint32_t u[8]; };

#if __has_builtin(__builtin_amdgcn_global_load_async_to_lds_b128) && \
    __has_builtin(__builtin_amdgcn_s_wait_asynccnt)
#define HAVE_ASYNC_LDS 1
typedef int v4i_ __attribute__((vector_size(16)));
#define GAS(p) ((__attribute__((address_space(1))) v4i_*)(p))
#define LAS(p) ((__attribute__((address_space(3))) v4i_*)(p))
#endif

// ---------------- f32 -> f16 convert + transpose (weights: KxN -> NxK) -----
__global__ void cvt_f16_t_kernel(const float* __restrict__ in, f16* __restrict__ outT,
                                 int K, int N) {
  int i = blockIdx.x * 256 + threadIdx.x;
  if (i < K * N) {
    int k = i / N, n = i - k * N;
    outT[(long)n * K + k] = (f16)in[i];
  }
}

// ---------------- L2 normalize rows of 320 (one wave per row) ----------------
__global__ __launch_bounds__(256) void l2norm_kernel(const float* __restrict__ x,
                                                     float* __restrict__ xn,
                                                     f16*   __restrict__ xh) {
  const int  lane = threadIdx.x & 31;
  const long row  = (long)blockIdx.x * 8 + (threadIdx.x >> 5);
  const float* p  = x + row * DIM;
  float v[10]; float ss = 0.f;
#pragma unroll
  for (int i = 0; i < 10; ++i) { v[i] = p[lane + i * 32]; ss += v[i] * v[i]; }
#pragma unroll
  for (int off = 16; off > 0; off >>= 1) ss += __shfl_xor(ss, off, 32);
  const float inv = 1.0f / fmaxf(sqrtf(ss), 1e-12f);
#pragma unroll
  for (int i = 0; i < 10; ++i) {
    float o = v[i] * inv;
    xn[row * DIM + lane + i * 32] = o;
    xh[row * DIM + lane + i * 32] = (f16)o;
  }
}

// ---------------- WMMA f16 GEMM: C[M,N] = A[M,K] @ BT[N,K]^T (+epilogue) ----
// Block tile 256x64, 8 waves; each wave: 32x64 strip = 2 A-frags x 4 B-frags
// = 8 v_wmma_f32_16x16x32_f16 per k-step.  BT is the pre-transposed weight.
// Double-buffered LDS: async tile(i+1) copies overlap tile(i) WMMA compute.
// MODE 0: out_h = f16(acc + bias)                       (qkv)
// MODE 1: out_f = acc + bias + res                      (proj + residual)
// MODE 2: out_h = f16(relu(acc))                        (mlp fc1)
// MODE 3: out_f = res + relu(acc + bias) * gamma        (mlp fc2 + layerscale)
template <int MODE>
__global__ __launch_bounds__(256) void gemm_kernel(
    const f16* __restrict__ A, const f16* __restrict__ BT,
    const float* __restrict__ bias, const float* __restrict__ res,
    const float* __restrict__ gamma,
    float* __restrict__ outF, f16* __restrict__ outH,
    int K, int N) {
  constexpr int BM = 256, BN = 64, BK = 32;
  __shared__ f16 As[2][BM][BK];   // 2 x 16 KB
  __shared__ f16 Bs[2][BN][BK];   // 2 x  4 KB

  const int tid  = threadIdx.x;
  const int lane = tid & 31;
  const int wave = tid >> 5;
  const int m16  = lane & 15;
  const int half = lane >> 4;
  const long mBase = (long)blockIdx.y * BM;
  const int  nBase = blockIdx.x * BN;

  // Stage one 256x32 A tile + 64x32 B tile into LDS buffer sb (16B chunks).
  auto stage = [&](int sb, int kb) {
    const f16* Ag = A  + mBase * K + kb;
    const f16* Bg = BT + (long)nBase * K + kb;
#if defined(HAVE_ASYNC_LDS) && defined(__HIP_DEVICE_COMPILE__)
#pragma unroll
    for (int i = 0; i < 4; ++i) {
      int c  = tid + i * 256;          // 16-byte chunk id, 0..1023
      int rr = c >> 2, kc = (c & 3) * 8;
      __builtin_amdgcn_global_load_async_to_lds_b128(
          GAS(Ag + (long)rr * K + kc), LAS(&As[sb][rr][kc]), 0, 0);
    }
    {
      int nn = tid >> 2, kc = (tid & 3) * 8;
      __builtin_amdgcn_global_load_async_to_lds_b128(
          GAS(Bg + (long)nn * K + kc), LAS(&Bs[sb][nn][kc]), 0, 0);
    }
#else
    uint4 ta[4];
#pragma unroll
    for (int i = 0; i < 4; ++i) {
      int c = tid + i * 256; int rr = c >> 2, kc = (c & 3) * 8;
      ta[i] = *(const uint4*)(Ag + (long)rr * K + kc);
    }
    uint4 tb;
    { int nn = tid >> 2, kc = (tid & 3) * 8;
      tb = *(const uint4*)(Bg + (long)nn * K + kc); }
#pragma unroll
    for (int i = 0; i < 4; ++i) {
      int c = tid + i * 256; int rr = c >> 2, kc = (c & 3) * 8;
      *(uint4*)&As[sb][rr][kc] = ta[i];
    }
    { int nn = tid >> 2, kc = (tid & 3) * 8;
      *(uint4*)&Bs[sb][nn][kc] = tb; }
#endif
  };

  v8f acc[2][4] = {};
  const int nk = K / BK;

  stage(0, 0);                               // prologue: tile 0 in flight
  for (int ik = 0; ik < nk; ++ik) {
    const int sb = ik & 1;
#if defined(HAVE_ASYNC_LDS) && defined(__HIP_DEVICE_COMPILE__)
    __builtin_amdgcn_s_wait_asynccnt(0);     // tile ik landed (issued last iter)
#endif
    __syncthreads();                         // all waves: tile ik visible,
                                             // buffer sb^1 no longer being read
    if (ik + 1 < nk) stage(sb ^ 1, (ik + 1) * BK);   // overlap with WMMAs below

    // ---- A fragments: 16-bit A 16x32 layout (ISA 7.12.2) ----
    Frag af[2];
#pragma unroll
    for (int u = 0; u < 2; ++u) {
      const int mrow = wave * 32 + u * 16 + m16;
#pragma unroll
      for (int i = 0; i < 8; ++i) {
        int k = ((i & 3) << 1) + (half << 3) + ((i >> 2) << 4);
        af[u].u[i] = *(const uint32_t*)&As[sb][mrow][k];
      }
    }
#pragma unroll
    for (int t = 0; t < 4; ++t) {
      // B fragment: lanes0-15 K=0..7,16..23, lanes16-31 K=8..15,24..31, N = lane&15
      Frag bf;
#pragma unroll
      for (int i = 0; i < 8; ++i) {
        int k = ((i & 3) << 1) + (half << 3) + ((i >> 2) << 4);
        bf.u[i] = *(const uint32_t*)&Bs[sb][t * 16 + m16][k];
      }
      acc[0][t] = __builtin_amdgcn_wmma_f32_16x16x32_f16(
          false, af[0].h, false, bf.h, (short)0, acc[0][t], false, false);
      acc[1][t] = __builtin_amdgcn_wmma_f32_16x16x32_f16(
          false, af[1].h, false, bf.h, (short)0, acc[1][t], false, false);
    }
  }

  // ---- epilogue: C layout VGPR r -> (m = r + 8*half, n = lane&15) ----
#pragma unroll
  for (int u = 0; u < 2; ++u) {
#pragma unroll
    for (int t = 0; t < 4; ++t) {
      const int n = nBase + t * 16 + m16;
#pragma unroll
      for (int r = 0; r < 8; ++r) {
        const long m   = mBase + wave * 32 + u * 16 + r + half * 8;
        const long idx = m * (long)N + n;
        float v = acc[u][t][r];
        if (MODE == 0)      outH[idx] = (f16)(v + bias[n]);
        else if (MODE == 1) outF[idx] = v + bias[n] + res[idx];
        else if (MODE == 2) outH[idx] = (f16)fmaxf(v, 0.f);
        else                outF[idx] = res[idx] + fmaxf(v + bias[n], 0.f) * gamma[n];
      }
    }
  }
}

// ---------------- windowed attention: one wave per (window, head) ----------
__global__ __launch_bounds__(256) void attn_kernel(const f16* __restrict__ qkv,
                                                   f16* __restrict__ ao) {
  const int waveId = threadIdx.x >> 5;
  const int lane   = threadIdx.x & 31;
  const int gw     = blockIdx.x * 8 + waveId;   // 0 .. 65535
  const int w      = gw >> 5;                   // window 0..2047
  const int h      = gw & 31;                   // head
  const int b  = w >> 8;
  const int hb = (w >> 4) & 15;
  const int wb = w & 15;

  __shared__ float sq[8][64][DH];
  __shared__ float sk[8][64][DH];
  __shared__ float sv[8][64][DH];

#pragma unroll
  for (int pass = 0; pass < 2; ++pass) {
    int t  = lane + pass * 32;
    int r  = t >> 3, cc = t & 7;
    long p = ((long)(b * 128 + hb * 8 + r)) * 128 + (wb * 8 + cc);
    const f16* src = qkv + p * 960 + h * 30;
#pragma unroll
    for (int j = 0; j < DH; ++j) {
      sq[waveId][t][j] = (float)src[j];
      sk[waveId][t][j] = (float)src[DH + j];
      sv[waveId][t][j] = (float)src[2 * DH + j];
    }
  }
  __syncthreads();

  const float scale = 0.31622776601683794f;  // 10^-0.5
#pragma unroll
  for (int pass = 0; pass < 2; ++pass) {
    const int i = lane + pass * 32;
    float qv[DH];
#pragma unroll
    for (int d = 0; d < DH; ++d) qv[d] = sq[waveId][i][d];
    float o[DH] = {};
    float mrun = -3.0e38f, lrun = 0.f;
    for (int j = 0; j < 64; ++j) {
      float sc = 0.f;
#pragma unroll
      for (int d = 0; d < DH; ++d) sc += qv[d] * sk[waveId][j][d];
      sc *= scale;
      float mnew = fmaxf(mrun, sc);
      float corr = __expf(mrun - mnew);
      float pj   = __expf(sc - mnew);
      lrun = lrun * corr + pj;
#pragma unroll
      for (int d = 0; d < DH; ++d) o[d] = o[d] * corr + pj * sv[waveId][j][d];
      mrun = mnew;
    }
    const float inv = 1.0f / lrun;
    int  r  = i >> 3, cc = i & 7;
    long p  = ((long)(b * 128 + hb * 8 + r)) * 128 + (wb * 8 + cc);
    f16* dst = ao + p * DIM + h * DH;
#pragma unroll
    for (int d = 0; d < DH; ++d) dst[d] = (f16)(o[d] * inv);
  }
}

// ---------------- launch ----------------
extern "C" void kernel_launch(void* const* d_in, const int* in_sizes, int n_in,
                              void* d_out, int out_size, void* d_ws, size_t ws_size,
                              hipStream_t stream) {
  (void)in_sizes; (void)n_in; (void)out_size; (void)ws_size;
  const float* x      = (const float*)d_in[0];
  const float* qkv_w  = (const float*)d_in[1];
  const float* qkv_b  = (const float*)d_in[2];
  const float* proj_w = (const float*)d_in[3];
  const float* proj_b = (const float*)d_in[4];
  const float* gamma  = (const float*)d_in[5];
  const float* w1     = (const float*)d_in[6];
  const float* w2     = (const float*)d_in[7];
  const float* b2     = (const float*)d_in[8];
  float* out = (float*)d_out;

  char* ws = (char*)d_ws;
  size_t off = 0;
  auto alloc = [&](size_t bytes) -> void* {
    void* p = ws + off;
    off = (off + bytes + 255) & ~(size_t)255;
    return p;
  };
  const long M = (long)MTOK;
  float* xn_f    = (float*)alloc(M * DIM * 4);
  f16*   xn_h    = (f16*)  alloc(M * DIM * 2);
  f16*   qkv_h   = (f16*)  alloc(M * 960 * 2);
  f16*   ao_h    = (f16*)  alloc(M * DIM * 2);
  float* y_f     = (float*)alloc(M * DIM * 4);
  float* x2_f    = (float*)alloc(M * DIM * 4);
  f16*   x2_h    = (f16*)  alloc(M * DIM * 2);
  f16*   h_h     = (f16*)  alloc(M * 1280 * 2);
  f16*   wqkvT   = (f16*)  alloc(960 * 320 * 2);   // [N=960][K=320]
  f16*   wprojT  = (f16*)  alloc(320 * 320 * 2);   // [320][320]
  f16*   w1T     = (f16*)  alloc(1280 * 320 * 2);  // [1280][320]
  f16*   w2T     = (f16*)  alloc(320 * 1280 * 2);  // [320][1280]

  cvt_f16_t_kernel<<<(320 * 960 + 255) / 256, 256, 0, stream>>>(qkv_w, wqkvT, 320, 960);
  cvt_f16_t_kernel<<<(320 * 320 + 255) / 256, 256, 0, stream>>>(proj_w, wprojT, 320, 320);
  cvt_f16_t_kernel<<<(320 * 1280 + 255) / 256, 256, 0, stream>>>(w1, w1T, 320, 1280);
  cvt_f16_t_kernel<<<(1280 * 320 + 255) / 256, 256, 0, stream>>>(w2, w2T, 1280, 320);

  // 1) x_norm = l2norm(x)
  l2norm_kernel<<<MTOK / 8, 256, 0, stream>>>(x, xn_f, xn_h);
  // 2) qkv = x_norm @ qkv_w + qkv_b            (M x 960)
  gemm_kernel<0><<<dim3(960 / 64, MTOK / 256), 256, 0, stream>>>(
      xn_h, wqkvT, qkv_b, nullptr, nullptr, nullptr, qkv_h, 320, 960);
  // 3) windowed multi-head attention           (M x 320)
  attn_kernel<<<8192, 256, 0, stream>>>(qkv_h, ao_h);
  // 4) y = attn_out @ proj_w + proj_b + x_norm (residual)
  gemm_kernel<1><<<dim3(320 / 64, MTOK / 256), 256, 0, stream>>>(
      ao_h, wprojT, proj_b, xn_f, nullptr, y_f, nullptr, 320, 320);
  // 5) x2 = l2norm(y)
  l2norm_kernel<<<MTOK / 8, 256, 0, stream>>>(y_f, x2_f, x2_h);
  // 6) h = relu(x2 @ w1)                       (M x 1280)
  gemm_kernel<2><<<dim3(1280 / 64, MTOK / 256), 256, 0, stream>>>(
      x2_h, w1T, nullptr, nullptr, nullptr, nullptr, h_h, 320, 1280);
  // 7) out = x2 + relu(h @ w2 + b2) * gamma
  gemm_kernel<3><<<dim3(320 / 64, MTOK / 256), 256, 0, stream>>>(
      h_h, w2T, b2, x2_f, gamma, out, nullptr, 1280, 320);
}
